// MultiHeadSelfAttention_33569464386001
// MI455X (gfx1250) — matrix operand order repaired and checked by reading, not compiled, not run on hardware
//
#include <hip/hip_runtime.h>
#include <math.h>

// ---------------------------------------------------------------------------
// MHA: B=4, S=2048, D=512, H=8, DK=64.  fp32 in/out, bf16 WMMA internally.
//   1) cvt x, W_Q/K/V -> bf16 (workspace)
//   2) proj: Q = (x Wq^T)*scale, K = x Wk^T  stored [b][h][s][dk] bf16
//            V = x Wv^T stored transposed   [b][h][dk][s] bf16
//      (one wave = 32x64 output tile: each weight fragment feeds 2 WMMAs)
//   3) fused attention, one wave = 32 queries (two 16-row tiles):
//      each K/V fragment feeds 2 WMMAs -> 2x operand reuse + 2x ILP per wait.
// ---------------------------------------------------------------------------

#define BB 4
#define SS 2048
#define DD 512
#define HH 8
#define DK 64
#define SCALE 0.125f   // 1/sqrt(64)

typedef __attribute__((ext_vector_type(16))) __bf16        v16bf;
typedef __attribute__((ext_vector_type(8)))  float         v8f;
typedef __attribute__((ext_vector_type(4)))  unsigned int  v4u;
typedef __attribute__((ext_vector_type(4)))  float         v4f;
typedef __attribute__((ext_vector_type(2)))  unsigned int  v2u;

union Frag16 { v16bf v; v4u u[2]; };

__device__ __forceinline__ unsigned short f32_to_bf16(float f) {
  union { float f; unsigned int u; } x; x.f = f;
  unsigned int u = x.u;
  u += 0x7FFFu + ((u >> 16) & 1u);   // round-to-nearest-even
  return (unsigned short)(u >> 16);
}

__device__ __forceinline__ unsigned int pack_bf16x2(float lo, float hi) {
  return (unsigned int)f32_to_bf16(lo) | ((unsigned int)f32_to_bf16(hi) << 16);
}

__device__ __forceinline__ v8f wmma_bf16(const Frag16& a, const Frag16& b, v8f c) {
  return __builtin_amdgcn_wmma_f32_16x16x32_bf16(
      /*neg_a=*/false, a.v, /*neg_b=*/false, b.v,
      /*c_mod=*/(short)0, c, /*reuse_a=*/false, /*reuse_b=*/false);
}

// ------------------------- fp32 -> bf16 conversion -------------------------
__global__ void __launch_bounds__(256)
cvt_bf16_kernel(const float* __restrict__ src, unsigned short* __restrict__ dst, int n4) {
  int i = blockIdx.x * blockDim.x + threadIdx.x;
  if (i >= n4) return;
  v4f v = ((const v4f*)src)[i];
  v2u o;
  o.x = pack_bf16x2(v.x, v.y);
  o.y = pack_bf16x2(v.z, v.w);
  ((v2u*)dst)[i] = o;
}

// ------------------------------ projections --------------------------------
// One wave computes a 32(m) x 64(n) tile of y = x @ W^T for one weight.
// The 4 weight B-fragments per k-chunk are reused by both 16-row A tiles.
__global__ void __launch_bounds__(256)
proj_kernel(const unsigned short* __restrict__ xb,
            const unsigned short* __restrict__ wb,
            unsigned short* __restrict__ qb,
            unsigned short* __restrict__ kb,
            unsigned short* __restrict__ vtb) {
  const int lane = threadIdx.x & 31;
  const int wave = threadIdx.x >> 5;
  const int gw   = blockIdx.x * 8 + wave;       // 0..6143
  const int widx = gw >> 11;                    // 0:Q 1:K 2:V   (2048 each)
  const int tid  = gw & 2047;
  const int m0   = (tid >> 3) * 32;             // 256 row-pair tiles
  const int h    = tid & 7;                     // n-tile == head
  const int n0   = h * 64;
  const int ln   = lane & 15;
  const int half = lane >> 4;

  const unsigned short* W = wb + (size_t)widx * DD * DD;
  const unsigned short* arow0 = xb + (size_t)(m0 + ln) * DD;
  const unsigned short* arow1 = arow0 + (size_t)16 * DD;

  v8f acc[2][4];
#pragma unroll
  for (int mt = 0; mt < 2; ++mt)
#pragma unroll
    for (int j = 0; j < 4; ++j) acc[mt][j] = v8f{};

  for (int k0 = 0; k0 < DD; k0 += 32) {
    Frag16 a0, a1;
    a0.u[0] = *(const v4u*)(arow0 + k0 + half * 8);
    a0.u[1] = *(const v4u*)(arow0 + k0 + 16 + half * 8);
    a1.u[0] = *(const v4u*)(arow1 + k0 + half * 8);
    a1.u[1] = *(const v4u*)(arow1 + k0 + 16 + half * 8);
#pragma unroll
    for (int j = 0; j < 4; ++j) {
      const unsigned short* brow = W + (size_t)(n0 + j * 16 + ln) * DD + k0 + half * 8;
      Frag16 b;
      b.u[0] = *(const v4u*)(brow);
      b.u[1] = *(const v4u*)(brow + 16);
      acc[0][j] = wmma_bf16(a0, b, acc[0][j]);   // same B, two independent
      acc[1][j] = wmma_bf16(a1, b, acc[1][j]);   // WMMA chains
    }
  }

  const int b  = m0 >> 11;          // batch (S=2048 rows; 32 | 2048)
  const int s0 = m0 & 2047;
  const size_t bh = (size_t)(b * HH + h);

#pragma unroll
  for (int mt = 0; mt < 2; ++mt) {
#pragma unroll
    for (int j = 0; j < 4; ++j) {
#pragma unroll
      for (int r = 0; r < 8; ++r) {
        const int m  = r + 8 * half;          // C-layout: row within tile
        const int s  = s0 + mt * 16 + m;
        const int dk = j * 16 + ln;           // C-layout: col = lane%16
        const float val = acc[mt][j][r];
        if (widx == 0) {
          qb[(bh * SS + s) * DK + dk] = f32_to_bf16(val * SCALE);
        } else if (widx == 1) {
          kb[(bh * SS + s) * DK + dk] = f32_to_bf16(val);
        } else {
          vtb[(bh * DK + dk) * SS + s] = f32_to_bf16(val);   // transposed V
        }
      }
    }
  }
}

// --------------------------- fused attention -------------------------------
// One wave owns two 16-query tiles (32 queries) of one (b,h); streams all
// 2048 keys in chunks of 32.  Every K/V fragment feeds two WMMAs.
__global__ void __launch_bounds__(256)
attn_kernel(const unsigned short* __restrict__ qb,
            const unsigned short* __restrict__ kb,
            const unsigned short* __restrict__ vtb,
            float* __restrict__ out) {
  __shared__ __align__(16) unsigned short lds_p[8][2][16][32]; // wave, qtile

  const int lane  = threadIdx.x & 31;
  const int wave  = threadIdx.x >> 5;
  const int gw    = blockIdx.x * 8 + wave;    // 0..2047
  const int bh    = gw >> 6;                  // 0..31
  const int qp    = gw & 63;                  // query-pair tile
  const int b     = bh >> 3;
  const int h     = bh & 7;
  const int ln    = lane & 15;
  const int half  = lane >> 4;
  const int q0    = qp * 32;

  const unsigned short* Q  = qb  + (size_t)bh * SS * DK;
  const unsigned short* K  = kb  + (size_t)bh * SS * DK;
  const unsigned short* Vt = vtb + (size_t)bh * DK * SS;

  // Q A-fragments: [qtile][dk-chunk]  (scale already folded in)
  Frag16 aq[2][2];
#pragma unroll
  for (int qt = 0; qt < 2; ++qt) {
    const unsigned short* qrow = Q + (size_t)(q0 + qt * 16 + ln) * DK;
    aq[qt][0].u[0] = *(const v4u*)(qrow + half * 8);
    aq[qt][0].u[1] = *(const v4u*)(qrow + 16 + half * 8);
    aq[qt][1].u[0] = *(const v4u*)(qrow + 32 + half * 8);
    aq[qt][1].u[1] = *(const v4u*)(qrow + 48 + half * 8);
  }

  v8f ctx[2][4];
#pragma unroll
  for (int qt = 0; qt < 2; ++qt)
#pragma unroll
    for (int j = 0; j < 4; ++j) ctx[qt][j] = v8f{};
  float rs[2][8] = {};

  for (int t0 = 0; t0 < SS; t0 += 32) {
    // ---- scores: two 16-key tiles; each K fragment used by both q-tiles
#pragma unroll
    for (int tt = 0; tt < 2; ++tt) {
      const unsigned short* kr = K + (size_t)(t0 + tt * 16 + ln) * DK + half * 8;
      Frag16 b0, b1;
      b0.u[0] = *(const v4u*)(kr);
      b0.u[1] = *(const v4u*)(kr + 16);
      b1.u[0] = *(const v4u*)(kr + 32);
      b1.u[1] = *(const v4u*)(kr + 48);

      v8f c0 = {}, c1 = {};
      c0 = wmma_bf16(aq[0][0], b0, c0);
      c1 = wmma_bf16(aq[1][0], b0, c1);
      c0 = wmma_bf16(aq[0][1], b1, c0);
      c1 = wmma_bf16(aq[1][1], b1, c1);

      // exp (reference: no max subtraction), rowsum, stage to LDS
#pragma unroll
      for (int r = 0; r < 8; ++r) {
        const float e0 = __expf(c0[r]);
        const float e1 = __expf(c1[r]);
        rs[0][r] += e0;
        rs[1][r] += e1;
        lds_p[wave][0][r + 8 * half][tt * 16 + ln] = f32_to_bf16(e0);
        lds_p[wave][1][r + 8 * half][tt * 16 + ln] = f32_to_bf16(e1);
      }
    }

    // intra-wave DS RAW: b16 scatter -> b128 re-layout gather
    asm volatile("s_wait_dscnt 0" ::: "memory");

    // ---- P as A-fragments (16 x 32, k = key index), one per q-tile
    Frag16 ap[2];
#pragma unroll
    for (int qt = 0; qt < 2; ++qt) {
      ap[qt].u[0] = *(const v4u*)&lds_p[wave][qt][ln][half * 8];
      ap[qt].u[1] = *(const v4u*)&lds_p[wave][qt][ln][16 + half * 8];
    }

    // ---- ctx += P @ V; each V fragment feeds both q-tiles
#pragma unroll
    for (int j = 0; j < 4; ++j) {
      const unsigned short* vrow = Vt + (size_t)(j * 16 + ln) * SS + t0 + half * 8;
      Frag16 bv;
      bv.u[0] = *(const v4u*)(vrow);
      bv.u[1] = *(const v4u*)(vrow + 16);
      ctx[0][j] = wmma_bf16(ap[0], bv, ctx[0][j]);
      ctx[1][j] = wmma_bf16(ap[1], bv, ctx[1][j]);
    }
  }

  // ---- reduce rowsums across the 16-lane half-group (wave32 shuffles)
#pragma unroll
  for (int qt = 0; qt < 2; ++qt) {
#pragma unroll
    for (int r = 0; r < 8; ++r) {
      float v = rs[qt][r];
      v += __shfl_xor(v, 1, 32);
      v += __shfl_xor(v, 2, 32);
      v += __shfl_xor(v, 4, 32);
      v += __shfl_xor(v, 8, 32);
      rs[qt][r] = v + 1e-8f;
    }
  }

  // ---- normalize + store fp32 out[b][s][h*64+dk]
#pragma unroll
  for (int qt = 0; qt < 2; ++qt) {
    float* ob = out + ((size_t)b * SS + q0 + qt * 16) * DD + h * DK;
#pragma unroll
    for (int j = 0; j < 4; ++j) {
#pragma unroll
      for (int r = 0; r < 8; ++r) {
        const int m = r + 8 * half;
        ob[(size_t)m * DD + j * 16 + ln] = ctx[qt][j][r] / rs[qt][r];
      }
    }
  }
}

// ------------------------------- launcher ----------------------------------
extern "C" void kernel_launch(void* const* d_in, const int* in_sizes, int n_in,
                              void* d_out, int out_size, void* d_ws, size_t ws_size,
                              hipStream_t stream) {
  const float* x  = (const float*)d_in[0];
  const float* Wq = (const float*)d_in[1];
  const float* Wk = (const float*)d_in[2];
  const float* Wv = (const float*)d_in[3];
  float* out = (float*)d_out;

  char* ws = (char*)d_ws;
  const size_t XB_BYTES = (size_t)BB * SS * DD * 2;        // 8 MB
  const size_t WB_BYTES = (size_t)3 * DD * DD * 2;         // 1.5 MB
  const size_t QK_BYTES = (size_t)BB * HH * SS * DK * 2;   // 8 MB each
  unsigned short* xb  = (unsigned short*)(ws);
  unsigned short* wb  = (unsigned short*)(ws + XB_BYTES);
  unsigned short* qb  = (unsigned short*)(ws + XB_BYTES + WB_BYTES);
  unsigned short* kb  = (unsigned short*)(ws + XB_BYTES + WB_BYTES + QK_BYTES);
  unsigned short* vtb = (unsigned short*)(ws + XB_BYTES + WB_BYTES + 2 * QK_BYTES);

  // 1) convert inputs to bf16
  {
    const int n4x = BB * SS * DD / 4;                      // 1,048,576
    cvt_bf16_kernel<<<n4x / 256, 256, 0, stream>>>(x, xb, n4x);
    const int n4w = DD * DD / 4;                           // 65,536
    cvt_bf16_kernel<<<n4w / 256, 256, 0, stream>>>(Wq, wb,               n4w);
    cvt_bf16_kernel<<<n4w / 256, 256, 0, stream>>>(Wk, wb + DD * DD,     n4w);
    cvt_bf16_kernel<<<n4w / 256, 256, 0, stream>>>(Wv, wb + 2 * DD * DD, n4w);
  }

  // 2) QKV projections: 3 * 2048 wave-tiles (32x64 each), 8 waves/block
  proj_kernel<<<768, 256, 0, stream>>>(xb, wb, qb, kb, vtb);

  // 3) fused attention: 2048 query-pair waves, 8 waves/block
  attn_kernel<<<256, 256, 0, stream>>>(qb, kb, vtb, out);
}
